// ImpactPredictor_60189671686204
// MI455X (gfx1250) — matrix operand order repaired and checked by reading, not compiled, not run on hardware
//
#include <hip/hip_runtime.h>
#include <hip/hip_bf16.h>

// Problem constants (from the reference)
#define N_NODES 50000
#define N_EDGES 800000
#define NHEADS  4
#define HIDCH   64
#define HC      256   // NHEADS*HIDCH
#define OUTD    32
#define MAX_KTILES 8  // K <= 256

typedef unsigned short u16;
typedef unsigned int   u32;
typedef __attribute__((ext_vector_type(16))) __bf16 v16bf;
typedef __attribute__((ext_vector_type(8)))  float  v8f;

// ---------------- helpers ----------------
__device__ __forceinline__ u16 f32_to_bf16_rte(float f) {
  u32 u = __builtin_bit_cast(u32, f);
  u32 r = 0x7FFFu + ((u >> 16) & 1u);     // round-to-nearest-even
  return (u16)((u + r) >> 16);
}
// ordered-uint encoding of float so unsigned atomicMax == float max
__device__ __forceinline__ u32 enc_f32(float f) {
  u32 u = __builtin_bit_cast(u32, f);
  return (u & 0x80000000u) ? ~u : (u | 0x80000000u);
}
__device__ __forceinline__ float dec_f32(u32 u) {
  u32 b = (u & 0x80000000u) ? (u & 0x7FFFFFFFu) : ~u;
  return __builtin_bit_cast(float, b);
}
#define ENC_NEG_INF 0x007FFFFFu   // enc_f32(-inf)

// ---------------- fills ----------------
__global__ void fill_f32(float* p, float v, int n) {
  int i = blockIdx.x * 256 + threadIdx.x;
  if (i < n) p[i] = v;
}
__global__ void fill_u32(u32* p, u32 v, int n) {
  int i = blockIdx.x * 256 + threadIdx.x;
  if (i < n) p[i] = v;
}

// ---------------- conversions ----------------
// Pack weights f32 [Kin x Nc] (zero-pad rows to Kpad) into WMMA-fragment-major
// bf16 layout: out[((nt*Ktiles + kt)*32 + lane)*16 + j], where lane holds
// column nt*16 + lane%16 and j indexes K = kt*32 + (lane&16 ? 16:0) + j.
// A lane's 16 B-operand values are then 32 contiguous bytes.
__global__ void pack_b_frag(const float* __restrict__ W, u16* __restrict__ out,
                            int Kin, int Kpad, int Nc) {
  int ktiles = Kpad >> 5;
  int total = (Nc >> 4) * ktiles * 512;
  int i = blockIdx.x * 256 + threadIdx.x;
  if (i >= total) return;
  int j    = i & 15;
  int lane = (i >> 4) & 31;
  int t    = i >> 9;
  int kt   = t % ktiles;
  int nt   = t / ktiles;
  int k = kt * 32 + ((lane & 16) ? 16 : 0) + j;
  int c = nt * 16 + (lane & 15);
  float v = (k < Kin) ? W[k * Nc + c] : 0.f;
  out[i] = f32_to_bf16_rte(v);
}
// f32 [R x Cin] -> bf16 [R x Cpad], zero-padded extra cols (x features, 7->32)
__global__ void cvt_pad_cols(const float* __restrict__ in, u16* __restrict__ out,
                             int R, int Cin, int Cpad) {
  long i = (long)blockIdx.x * 256 + threadIdx.x;
  if (i >= (long)R * Cpad) return;
  int r = (int)(i / Cpad), c = (int)(i - (long)r * Cpad);
  out[i] = f32_to_bf16_rte(c < Cin ? in[(long)r * Cin + c] : 0.f);
}

// ---------------- bf16 WMMA GEMM: C[MxN] = A[MxK] * B[KxN], f32 accumulate ----
// Block = 8 waves; block owns one n-tile and 8 consecutive m-tiles. The packed
// B panel for the n-tile is DMA'd once into LDS via global_load_async_to_lds_b128
// (ASYNCcnt), then every wave reads its fragment with wide LDS loads. One
// v_wmma_f32_16x16x32_bf16 per k-step per wave.
__global__ void gemm_bf16_wmma(const u16* __restrict__ A, const u16* __restrict__ Bp,
                               float* __restrict__ C, int M, int Nn, int K) {
  __shared__ __attribute__((aligned(16))) u16 sB[MAX_KTILES * 32 * 16]; // 8KB max
  const int ktiles = K >> 5;
  const int ntiles = Nn >> 4;
  const int mtiles = M >> 4;
  const int ntile  = blockIdx.x % ntiles;
  const int mbase  = (blockIdx.x / ntiles) * 8;
  const int waveid = threadIdx.x >> 5;
  const int lane   = threadIdx.x & 31;

  // ---- async-DMA the packed B panel for this n-tile into LDS ----
  {
    const u16* gB = Bp + (size_t)ntile * ktiles * 512;  // 512 u16 per ktile
    const int chunks = ktiles * 64;                     // 16-byte chunks
    for (int idx = threadIdx.x; idx < chunks; idx += 256) {
      u32 ldsoff = (u32)(uintptr_t)&sB[idx * 8];        // low 32b = LDS offset
      const u16* gp = gB + idx * 8;
      asm volatile("global_load_async_to_lds_b128 %0, %1, off"
                   :: "v"(ldsoff), "v"(gp) : "memory");
    }
    asm volatile("s_wait_asynccnt 0" ::: "memory");
  }
  __syncthreads();

  int mtile = mbase + waveid;
  const bool active = (mtile < mtiles);
  if (!active) mtile = mtiles - 1;          // clamp: keep EXEC full, skip store
  const int hi   = (lane >> 4) & 1;
  const int rowA = mtile * 16 + (lane & 15);   // A: M = lane%16
  const int colC = ntile * 16 + (lane & 15);   // B/C: N = lane%16

  v8f acc = {0.f, 0.f, 0.f, 0.f, 0.f, 0.f, 0.f, 0.f};
  const u32* pa = (const u32*)(A + (size_t)rowA * K);
  const u32* pb = (const u32*)sB;

  for (int kt = 0; kt < ktiles; ++kt) {
    const int k0 = kt * 32;
    // A fragment: regs 0..3 hold K = k0 + hi*16 + {0..7},
    //             regs 4..7 hold K = k0 + hi*16 + 8 + {0..7} (pairs per dword)
    union { v16bf v; u32 u[8]; } fa;
    const int ka = (k0 + (hi ? 16 : 0)) >> 1;
    const int kb = (k0 + 8 + (hi ? 16 : 0)) >> 1;
#pragma unroll
    for (int r = 0; r < 4; ++r) { fa.u[r] = pa[ka + r]; fa.u[4 + r] = pa[kb + r]; }

    // B fragment: 32 contiguous bytes per lane from LDS
    union { v16bf v; u32 u[8]; } fb;
    const u32* fp = pb + (size_t)(kt * 32 + lane) * 8;
#pragma unroll
    for (int r = 0; r < 8; ++r) fb.u[r] = fp[r];

    if (kt + 1 < ktiles) __builtin_prefetch(A + (size_t)rowA * K + k0 + 32, 0, 1);

    acc = __builtin_amdgcn_wmma_f32_16x16x32_bf16(
        /*neg_a=*/false, fa.v, /*neg_b=*/false, fb.v,
        /*c_mod=*/(short)0, acc, /*reuse_a=*/false, /*reuse_b=*/false);
  }

  if (active) {
    // C store: VGPR r -> M = mtile*16 + (hi?8:0) + r, N = ntile*16 + lane%16
    const int rbase = mtile * 16 + (hi ? 8 : 0);
#pragma unroll
    for (int r = 0; r < 8; ++r) C[(size_t)(rbase + r) * Nn + colC] = acc[r];
  }
}

// ---------------- GAT attention ----------------
// per-node attention logits: es[n,h] = <h[n,h,:], a_s[h,:]>, ed likewise
__global__ void attn_logits(const float* __restrict__ h, const float* __restrict__ a_s,
                            const float* __restrict__ a_d, float* __restrict__ es,
                            float* __restrict__ ed, int Nn, int NH, int CH) {
  int i = blockIdx.x * 256 + threadIdx.x;
  if (i >= Nn * NH) return;
  int n = i / NH, hh = i - n * NH;
  const float* hp = h + (size_t)n * NH * CH + (size_t)hh * CH;
  float s = 0.f, d = 0.f;
  for (int c = 0; c < CH; ++c) {
    s += hp[c] * a_s[hh * CH + c];
    d += hp[c] * a_d[hh * CH + c];
  }
  es[i] = s;
  ed[i] = d;
}

// pass 1: e = leaky_relu(es[src]+ed[dst]); stash e; segment max via atomic u32 max
__global__ void edge_max(const int* __restrict__ srcv, const int* __restrict__ dstv,
                         int E0, int Nn, const float* __restrict__ es,
                         const float* __restrict__ ed, float* __restrict__ ebuf,
                         u32* __restrict__ m, int NH) {
  int e = blockIdx.x * 256 + threadIdx.x;
  int Etot = E0 + Nn;
  if (e >= Etot) return;
  int s = (e < E0) ? srcv[e] : (e - E0);   // self-loops appended
  int d = (e < E0) ? dstv[e] : (e - E0);
  for (int hh = 0; hh < NH; ++hh) {
    float v = es[s * NH + hh] + ed[d * NH + hh];
    v = (v > 0.f) ? v : 0.2f * v;          // leaky_relu slope 0.2
    ebuf[(size_t)e * NH + hh] = v;
    atomicMax(&m[d * NH + hh], enc_f32(v));
  }
}

// pass 2: ex = exp(e - m[dst]); stash ex; segment sum via atomic f32 add
__global__ void edge_expsum(const int* __restrict__ srcv, const int* __restrict__ dstv,
                            int E0, int Nn, float* __restrict__ ebuf,
                            const u32* __restrict__ m, float* __restrict__ denom, int NH) {
  int e = blockIdx.x * 256 + threadIdx.x;
  int Etot = E0 + Nn;
  if (e >= Etot) return;
  int d = (e < E0) ? dstv[e] : (e - E0);
  for (int hh = 0; hh < NH; ++hh) {
    float ex = expf(ebuf[(size_t)e * NH + hh] - dec_f32(m[d * NH + hh]));
    ebuf[(size_t)e * NH + hh] = ex;
    atomicAdd(&denom[d * NH + hh], ex);
  }
}

// pass 3: out[dst] += alpha * h[src]  (one wave per edge; lane strides channels)
__global__ void edge_scatter(const int* __restrict__ srcv, const int* __restrict__ dstv,
                             int E0, int Nn, const float* __restrict__ ebuf,
                             const float* __restrict__ denom, const float* __restrict__ h,
                             float* __restrict__ outp, int NH, int CH) {
  int wave = blockIdx.x * (blockDim.x >> 5) + (threadIdx.x >> 5);
  int lane = threadIdx.x & 31;
  int Etot = E0 + Nn;
  if (wave >= Etot) return;
  int s = (wave < E0) ? srcv[wave] : (wave - E0);
  int d = (wave < E0) ? dstv[wave] : (wave - E0);
  int HCl = NH * CH;
  for (int c = lane; c < HCl; c += 32) {
    int hh = c / CH;
    float alpha = ebuf[(size_t)wave * NH + hh] / (denom[d * NH + hh] + 1e-16f);
    atomicAdd(&outp[(size_t)d * HCl + c], h[(size_t)s * HCl + c] * alpha);
  }
}

// ---------------- BN + bias ----------------
// add bias in-place, accumulate per-channel sum/sumsq (one atomic per block per ch)
__global__ void bias_stats(float* __restrict__ o, const float* __restrict__ bias,
                           float* __restrict__ sum, float* __restrict__ ssum, int Nn) {
  int c = threadIdx.x;          // 256 channels, blockDim.x == 256
  int r0 = blockIdx.x * 64;
  float s = 0.f, s2 = 0.f;
  for (int r = 0; r < 64; ++r) {
    int n = r0 + r;
    if (n >= Nn) break;
    float v = o[(size_t)n * HC + c] + bias[c];
    o[(size_t)n * HC + c] = v;
    s += v; s2 += v * v;
  }
  atomicAdd(&sum[c], s);
  atomicAdd(&ssum[c], s2);
}

// batchnorm + relu, emit bf16 activation for the next WMMA GEMM
__global__ void bn_relu_cvt(const float* __restrict__ o, const float* __restrict__ g,
                            const float* __restrict__ be, const float* __restrict__ sum,
                            const float* __restrict__ ssum, u16* __restrict__ out, int Nn) {
  long i = (long)blockIdx.x * 256 + threadIdx.x;
  if (i >= (long)Nn * HC) return;
  int c = (int)(i & (HC - 1));
  float mu  = sum[c] / (float)Nn;
  float var = ssum[c] / (float)Nn - mu * mu;
  float v = (o[i] - mu) * rsqrtf(var + 1e-5f) * g[c] + be[c];
  out[i] = f32_to_bf16_rte(fmaxf(v, 0.f));
}

// ---------------- MLP tail ----------------
__global__ void emb_bias(const float* __restrict__ o, const float* __restrict__ b3,
                         float* __restrict__ emb, int Nn) {
  long i = (long)blockIdx.x * 256 + threadIdx.x;
  if (i >= (long)Nn * OUTD) return;
  emb[i] = o[i] + b3[i & (OUTD - 1)];
}

// z[n] = concat(emb[source_node], emb[n]) as bf16 (source node read on device)
__global__ void build_z(const float* __restrict__ emb, const int* __restrict__ srcnode,
                        u16* __restrict__ zb, int Nn) {
  long i = (long)blockIdx.x * 256 + threadIdx.x;
  if (i >= (long)Nn * 64) return;
  int n = (int)(i >> 6), c = (int)(i & 63);
  float v = (c < OUTD) ? emb[(size_t)srcnode[0] * OUTD + c]
                       : emb[(size_t)n * OUTD + (c - OUTD)];
  zb[i] = f32_to_bf16_rte(v);
}

__global__ void relu_bias_cvt(const float* __restrict__ in, const float* __restrict__ bias,
                              u16* __restrict__ out, int R, int F) {
  long i = (long)blockIdx.x * 256 + threadIdx.x;
  if (i >= (long)R * F) return;
  int c = (int)(i % F);
  out[i] = f32_to_bf16_rte(fmaxf(in[i] + bias[c], 0.f));
}

// scores = sigmoid(relu(z2 + bm2) @ Wm3 + bm3)
__global__ void mlp_final(const float* __restrict__ z2, const float* __restrict__ bm2,
                          const float* __restrict__ Wm3, const float* __restrict__ bm3,
                          float* __restrict__ out, int Nn) {
  int n = blockIdx.x * 256 + threadIdx.x;
  if (n >= Nn) return;
  float s = bm3[0];
#pragma unroll
  for (int c = 0; c < 32; ++c) {
    float v = fmaxf(z2[(size_t)n * 32 + c] + bm2[c], 0.f);
    s += v * Wm3[c];
  }
  out[n] = 1.f / (1.f + expf(-s));
}

// ---------------- host orchestration ----------------
extern "C" void kernel_launch(void* const* d_in, const int* in_sizes, int n_in,
                              void* d_out, int out_size, void* d_ws, size_t ws_size,
                              hipStream_t stream) {
  (void)in_sizes; (void)n_in; (void)out_size; (void)ws_size;
  const float* x       = (const float*)d_in[0];
  const int*   ei      = (const int*)d_in[1];
  const int*   srcnode = (const int*)d_in[2];
  const float* W1  = (const float*)d_in[3];
  const float* as1 = (const float*)d_in[4];
  const float* ad1 = (const float*)d_in[5];
  const float* b1  = (const float*)d_in[6];
  const float* g1  = (const float*)d_in[7];
  const float* be1 = (const float*)d_in[8];
  const float* W2  = (const float*)d_in[9];
  const float* as2 = (const float*)d_in[10];
  const float* ad2 = (const float*)d_in[11];
  const float* b2  = (const float*)d_in[12];
  const float* g2  = (const float*)d_in[13];
  const float* be2 = (const float*)d_in[14];
  const float* W3  = (const float*)d_in[15];
  const float* as3 = (const float*)d_in[16];
  const float* ad3 = (const float*)d_in[17];
  const float* b3  = (const float*)d_in[18];
  const float* Wm1 = (const float*)d_in[19];
  const float* bm1 = (const float*)d_in[20];
  const float* Wm2 = (const float*)d_in[21];
  const float* bm2 = (const float*)d_in[22];
  const float* Wm3 = (const float*)d_in[23];
  const float* bm3 = (const float*)d_in[24];

  const int* esrcv = ei;            // edge_index[0]
  const int* edstv = ei + N_EDGES;  // edge_index[1]
  const int  Etot  = N_EDGES + N_NODES;

  // workspace carve-out (bump allocator, 256B aligned)
  char* base = (char*)d_ws;
  size_t off = 0;
  auto carve = [&](size_t bytes) -> char* {
    char* p = base + off;
    off = (off + bytes + 255) & ~(size_t)255;
    return p;
  };
  float* hbuf  = (float*)carve(sizeof(float) * (size_t)N_NODES * HC); // GEMM out
  float* obuf  = (float*)carve(sizeof(float) * (size_t)N_NODES * HC); // aggregated
  u16*   abuf  = (u16*)  carve(sizeof(u16)   * (size_t)N_NODES * HC); // bf16 acts
  u16*   wbuf  = (u16*)  carve(sizeof(u16)   * (size_t)HC * HC);      // packed weights
  float* ebuf  = (float*)carve(sizeof(float) * (size_t)Etot * NHEADS);
  float* esb   = (float*)carve(sizeof(float) * (size_t)N_NODES * NHEADS);
  float* edb   = (float*)carve(sizeof(float) * (size_t)N_NODES * NHEADS);
  u32*   mbuf  = (u32*)  carve(sizeof(u32)   * (size_t)N_NODES * NHEADS);
  float* dbuf  = (float*)carve(sizeof(float) * (size_t)N_NODES * NHEADS);
  float* embb  = (float*)carve(sizeof(float) * (size_t)N_NODES * OUTD);
  float* stats = (float*)carve(sizeof(float) * 512);

  dim3 blk(256);
  auto cdiv = [](long a, long b) -> int { return (int)((a + b - 1) / b); };

  auto packB = [&](const float* W, int Kin, int Kpad, int Nc) {
    pack_b_frag<<<cdiv((long)Kpad * Nc, 256), blk, 0, stream>>>(W, wbuf, Kin, Kpad, Nc);
  };
  auto gemm = [&](const u16* A, const u16* Bp, float* C, int M, int Nn, int K) {
    int ntiles = Nn / 16, mtiles = M / 16;
    int mblocks = (mtiles + 7) / 8;           // 8 waves (m-tiles) per block
    gemm_bf16_wmma<<<dim3(ntiles * mblocks), blk, 0, stream>>>(A, Bp, C, M, Nn, K);
  };

  auto attention = [&](const float* aS, const float* aD, int NH, int CH) {
    attn_logits<<<cdiv((long)N_NODES * NH, 256), blk, 0, stream>>>(hbuf, aS, aD, esb, edb, N_NODES, NH, CH);
    fill_f32<<<cdiv((long)N_NODES * NH * CH, 256), blk, 0, stream>>>(obuf, 0.f, N_NODES * NH * CH);
    fill_u32<<<cdiv((long)N_NODES * NH, 256), blk, 0, stream>>>(mbuf, ENC_NEG_INF, N_NODES * NH);
    fill_f32<<<cdiv((long)N_NODES * NH, 256), blk, 0, stream>>>(dbuf, 0.f, N_NODES * NH);
    edge_max<<<cdiv(Etot, 256), blk, 0, stream>>>(esrcv, edstv, N_EDGES, N_NODES, esb, edb, ebuf, mbuf, NH);
    edge_expsum<<<cdiv(Etot, 256), blk, 0, stream>>>(esrcv, edstv, N_EDGES, N_NODES, ebuf, mbuf, dbuf, NH);
    edge_scatter<<<cdiv((long)Etot * 32, 256), blk, 0, stream>>>(esrcv, edstv, N_EDGES, N_NODES,
                                                                 ebuf, dbuf, hbuf, obuf, NH, CH);
  };

  auto bn_block = [&](const float* bias, const float* g, const float* be) {
    fill_f32<<<cdiv(512, 256), blk, 0, stream>>>(stats, 0.f, 512);
    bias_stats<<<cdiv(N_NODES, 64), blk, 0, stream>>>(obuf, bias, stats, stats + 256, N_NODES);
    bn_relu_cvt<<<cdiv((long)N_NODES * HC, 256), blk, 0, stream>>>(obuf, g, be, stats, stats + 256, abuf, N_NODES);
  };

  // ---- GAT layer 1: x[50000x7] -> 256 ----
  cvt_pad_cols<<<cdiv((long)N_NODES * 32, 256), blk, 0, stream>>>(x, abuf, N_NODES, 7, 32);
  packB(W1, 7, 32, HC);
  gemm(abuf, wbuf, hbuf, N_NODES, HC, 32);
  attention(as1, ad1, NHEADS, HIDCH);
  bn_block(b1, g1, be1);

  // ---- GAT layer 2: 256 -> 256 ----
  packB(W2, HC, HC, HC);
  gemm(abuf, wbuf, hbuf, N_NODES, HC, HC);
  attention(as2, ad2, NHEADS, HIDCH);
  bn_block(b2, g2, be2);

  // ---- GAT layer 3: 256 -> 32, 1 head, mean==identity ----
  packB(W3, HC, HC, OUTD);
  gemm(abuf, wbuf, hbuf, N_NODES, OUTD, HC);
  attention(as3, ad3, 1, OUTD);
  emb_bias<<<cdiv((long)N_NODES * OUTD, 256), blk, 0, stream>>>(obuf, b3, embb, N_NODES);

  // ---- MLP head ----
  build_z<<<cdiv((long)N_NODES * 64, 256), blk, 0, stream>>>(embb, srcnode, abuf, N_NODES);
  packB(Wm1, 64, 64, 64);
  gemm(abuf, wbuf, hbuf, N_NODES, 64, 64);
  relu_bias_cvt<<<cdiv((long)N_NODES * 64, 256), blk, 0, stream>>>(hbuf, bm1, abuf, N_NODES, 64);
  packB(Wm2, 64, 64, 32);
  gemm(abuf, wbuf, obuf, N_NODES, 32, 64);
  mlp_final<<<cdiv(N_NODES, 256), blk, 0, stream>>>(obuf, bm2, Wm3, bm3, (float*)d_out, N_NODES);
}
